// StreamingDurationHead_26697516712416
// MI455X (gfx1250) — compile-verified, matrix operand order checked by reading, not run on hardware
//
#include <hip/hip_runtime.h>
#include <hip/hip_bf16.h>
#include <math.h>
#include <stdint.h>

// ---------------------------------------------------------------------------
// StreamingDurationHead for MI455X (gfx1250, wave32, WMMA).
// Heavy matmuls -> v_wmma_f32_16x16x32_bf16 with f32 accumulation.
// Weight k-slices staged in LDS with double-buffered async-to-LDS copies
// (ASYNCcnt / s_wait_asynccnt) so WMMAs never wait on global latency.
// ---------------------------------------------------------------------------

typedef __bf16 bf16_t;
typedef __attribute__((ext_vector_type(16))) __bf16 bf16x16;
typedef __attribute__((ext_vector_type(8)))  __bf16 bf16x8;
typedef __attribute__((ext_vector_type(8)))  float  f32x8;
typedef int v4i_gcc __attribute__((vector_size(16)));   // matches builtin param

#define B_   32
#define U_   2048
#define D_   256
#define M_   512
#define R_   (B_ * U_)          // 65536 rows
#define DECAY_ 0.95f
#define MAXLS_ 1.2f

// ---------------- CDNA5 async global->LDS copy (16 bytes / lane) -----------
__device__ __forceinline__ void async_ld_b128(uint32_t lds_addr, const void* gaddr) {
#if __has_builtin(__builtin_amdgcn_global_load_async_to_lds_b128)
  __builtin_amdgcn_global_load_async_to_lds_b128(
      (__attribute__((address_space(1))) v4i_gcc*)(unsigned long long)(uintptr_t)gaddr,
      (__attribute__((address_space(3))) v4i_gcc*)(uintptr_t)lds_addr,
      0, 0);
#else
  asm volatile("global_load_async_to_lds_b128 %0, %1, off"
               :: "v"(lds_addr), "v"(gaddr) : "memory");
#endif
}
__device__ __forceinline__ void wait_asynccnt0() {
  asm volatile("s_wait_asynccnt 0x0" ::: "memory");
}

// ---------------- WMMA fragment helpers (wave32) ---------------------------
// 16-bit A-matrix 16x32 layout (ISA 7.12.2): lane L holds row M=L%16,
// K-half = (L/16): VGPR0..3 = K kh..kh+7, VGPR4..7 = K 16+kh..16+kh+7.
// B fragments use the same per-lane layout with "row" = N (i.e. B^T packed).
__device__ __forceinline__ bf16x16 frag_zero() {
  bf16x16 z;
#pragma unroll
  for (int i = 0; i < 16; ++i) z[i] = (bf16_t)0.0f;
  return z;
}

__device__ __forceinline__ bf16x16 frag_load(const bf16_t* rowptr, int kofs, int lane) {
  const bf16_t* p = rowptr + kofs + ((lane >> 4) << 3);   // + khalf*8
  bf16x8 lo = *(const bf16x8*)p;                          // K kh..kh+7
  bf16x8 hi = *(const bf16x8*)(p + 16);                   // K 16+kh..16+kh+7
  bf16x16 r;
#pragma unroll
  for (int i = 0; i < 8; ++i) { r[i] = lo[i]; r[8 + i] = hi[i]; }
  return r;
}

__device__ __forceinline__ float wave_sum(float v) {
#pragma unroll
  for (int m = 16; m > 0; m >>= 1) v += __shfl_xor(v, m, 32);
  return v;
}
__device__ __forceinline__ float wave_max(float v) {
#pragma unroll
  for (int m = 16; m > 0; m >>= 1) v = fmaxf(v, __shfl_xor(v, m, 32));
  return v;
}

__device__ __forceinline__ float gelu_exact(float x) {
  return 0.5f * x * (1.0f + erff(x * 0.70710678118654752f));
}

// ---------------- K0: weight prepack ---------------------------------------
__global__ void k_prepack(const float* __restrict__ c1w, const float* __restrict__ c2w,
                          const float* __restrict__ ck,  const float* __restrict__ w1,
                          const float* __restrict__ cov,
                          bf16_t* p1, bf16_t* p2, bf16_t* pck, bf16_t* pw1,
                          float* lcov) {
  int i = blockIdx.x * blockDim.x + threadIdx.x;
  const int NC = D_ * D_ * 3;                       // 196608
  if (i < NC) {
    int h = i / (D_ * D_);
    int rem = i - h * D_ * D_;
    int o = rem >> 8, ii = rem & 255;
    p1[i] = (bf16_t)c1w[(o * D_ + ii) * 3 + h];
    p2[i] = (bf16_t)c2w[(o * D_ + ii) * 3 + h];
  }
  if (i < M_ * D_) pck[i] = (bf16_t)ck[i];          // already N x K
  if (i < D_ * 288) {
    int n = i / 288, k = i - n * 288;
    pw1[i] = (k < D_ + 4) ? (bf16_t)w1[k * D_ + n] : (bf16_t)0.0f;
  }
  if (i < B_ * M_) lcov[i] = logf(fmaxf(cov[i], 1e-4f));
}

// ---------------- K1: gated-EMA affine parallel scan + masks ---------------
__global__ void k_scan(const float* __restrict__ la, const float* __restrict__ um,
                       const float* __restrict__ sm, const float* __restrict__ sp,
                       const float* __restrict__ init,
                       float* maskc, float* speech, float* lrs) {
  int lane = threadIdx.x & 31;
  int b = blockIdx.x * (blockDim.x >> 5) + (threadIdx.x >> 5);
  if (b >= B_) return;
  const size_t base = (size_t)b * U_;
  const int u0 = lane * 64;

  float A = 1.0f, Bc = 0.0f;
  for (int j = 0; j < 64; ++j) {
    size_t r = base + u0 + j;
    float mk = fminf(fmaxf(um[r], 0.f), 1.f);
    float se = fminf(fmaxf(sm[r], 0.f), 1.f);
    float s  = fminf(fmaxf(sp[r], 0.f), 1.f);
    float m  = mk * se * (1.f - s);
    float a  = (m > 0.5f) ? DECAY_ : 1.0f;
    float bb = (m > 0.5f) ? (1.0f - DECAY_) * la[r] : 0.0f;
    A = a * A; Bc = a * Bc + bb;
  }
  float Ap = 1.0f, Bp = 0.0f;
  for (int j = 0; j < 32; ++j) {
    float Aj = __shfl(A, j, 32);
    float Bj = __shfl(Bc, j, 32);
    if (j < lane) { Bp = Aj * Bp + Bj; Ap = Aj * Ap; }
  }
  float prev = Ap * init[b] + Bp;

  for (int j = 0; j < 64; ++j) {
    size_t r = base + u0 + j;
    float mk = fminf(fmaxf(um[r], 0.f), 1.f);
    float se = fminf(fmaxf(sm[r], 0.f), 1.f);
    float s  = fminf(fmaxf(sp[r], 0.f), 1.f);
    float m  = mk * se * (1.f - s);
    maskc[r] = mk; speech[r] = m; lrs[r] = prev;
    if (m > 0.5f) prev = DECAY_ * prev + (1.0f - DECAY_) * la[r];
  }
}

// ---------------- K2: hidden0 = emb[cu] + aux @ aux_w + aux_b --------------
__global__ void k_hidden0(const int* __restrict__ cu, const float* __restrict__ la,
                          const float* __restrict__ lrs, const float* __restrict__ emb,
                          const float* __restrict__ aw, const float* __restrict__ ab,
                          bf16_t* __restrict__ h0) {
  int t = threadIdx.x;
  int r = blockIdx.x * 4 + (t >> 6);
  int d0 = (t & 63) * 4;
  int v = cu[r];
  float a0 = la[r], a3 = lrs[r];
  const float* e = emb + (size_t)v * D_ + d0;
#pragma unroll
  for (int j = 0; j < 4; ++j) {
    int d = d0 + j;
    float x = e[j] + a0 * aw[d] + a3 * aw[3 * D_ + d] + ab[d];
    h0[(size_t)r * D_ + d] = (bf16_t)x;
  }
}

// ---------------- K3/K4: causal conv as GEMMs, LDS-staged B ---------------
// 256 threads = 8 waves; tile 32 rows x 256 cols; wave = (row-tile, col-group).
// Per k-step the 256x32 bf16 weight slice (16KB) is async-copied into LDS,
// double-buffered; prefetch of step s+1 overlaps WMMAs of step s.
template <int TAPS>
__global__ __launch_bounds__(256) void k_gemm_conv(const bf16_t* __restrict__ A,
                                                   const bf16_t* __restrict__ Bp,
                                                   const float* __restrict__ bias,
                                                   bf16_t* __restrict__ Out) {
  __shared__ __align__(16) bf16_t sB[2][256 * 32];
  const int tid  = threadIdx.x;
  const int lane = tid & 31;
  const int wave = tid >> 5;
  const int rt   = wave >> 2;              // row tile 0/1
  const int cg   = wave & 3;               // col group -> n-tiles cg*4..cg*4+3
  const int row0 = blockIdx.x * 32;        // never crosses a batch boundary
  const int b    = row0 / U_;
  const int u0   = row0 - b * U_;
  const int m    = lane & 15;
  const int NSTEP = TAPS * 8;

  f32x8 acc[4] = {};
  const bf16_t* Abase = A + (size_t)b * U_ * D_;

  // thread tid copies weight row n=tid of the current k-slice (64 bytes)
  auto prefetch = [&](int s, int buf) {
    int h = s >> 3, kt = s & 7;
    const bf16_t* src = Bp + (size_t)h * D_ * D_ + (size_t)tid * D_ + kt * 32;
    uint32_t dst = (uint32_t)(uintptr_t)&sB[buf][tid * 32];
#pragma unroll
    for (int j = 0; j < 4; ++j) async_ld_b128(dst + j * 16, src + j * 8);
  };

  prefetch(0, 0);
  for (int s = 0; s < NSTEP; ++s) {
    wait_asynccnt0();                       // slice s landed (mine)
    __syncthreads();                        // everyone's landed
    if (s + 1 < NSTEP) prefetch(s + 1, (s + 1) & 1);

    const int h = s >> 3, kt = s & 7;
    const int shift = h - (TAPS - 1);       // -2,-1,0 causal taps
    const int up = u0 + rt * 16 + m + shift;
    bf16x16 afrag = (up >= 0) ? frag_load(Abase + (size_t)up * D_, kt * 32, lane)
                              : frag_zero();
    const bf16_t* sbase = sB[s & 1];
#pragma unroll
    for (int t = 0; t < 4; ++t) {
      bf16x16 bfrag = frag_load(sbase + (size_t)((cg * 4 + t) * 16 + m) * 32, 0, lane);
      acc[t] = __builtin_amdgcn_wmma_f32_16x16x32_bf16(
          false, afrag, false, bfrag, (short)0, acc[t], false, false);
    }
    __syncthreads();                        // reads done before buf reuse
  }

  const int rhi = (lane >> 4) * 8;          // C layout: M = v + 8*(lane/16)
#pragma unroll
  for (int t = 0; t < 4; ++t) {
    int col = (cg * 4 + t) * 16 + m;
    float bb = bias[col];
#pragma unroll
    for (int v = 0; v < 8; ++v) {
      int r = row0 + rt * 16 + rhi + v;
      float x = gelu_exact(acc[t][v] + bb);
      Out[(size_t)r * D_ + col] = (bf16_t)x;
    }
  }
}

// ---------------- K5: LayerNorm * mask -> query (bf16) ---------------------
__global__ void k_layernorm(const bf16_t* __restrict__ X, const float* __restrict__ g,
                            const float* __restrict__ bta, const float* __restrict__ maskc,
                            bf16_t* __restrict__ Q) {
  int lane = threadIdx.x & 31;
  int r = blockIdx.x * 8 + (threadIdx.x >> 5);
  const bf16_t* x = X + (size_t)r * D_;
  float vals[8];
#pragma unroll
  for (int j = 0; j < 8; ++j) vals[j] = (float)x[lane + 32 * j];
  float s = 0.f;
#pragma unroll
  for (int j = 0; j < 8; ++j) s += vals[j];
  float mu = wave_sum(s) * (1.0f / D_);
  float v2 = 0.f;
#pragma unroll
  for (int j = 0; j < 8; ++j) { float d = vals[j] - mu; v2 += d * d; }
  float inv = rsqrtf(wave_sum(v2) * (1.0f / D_) + 1e-5f);
  float mk = maskc[r];
#pragma unroll
  for (int j = 0; j < 8; ++j) {
    int c = lane + 32 * j;
    float y = (vals[j] - mu) * inv * g[c] + bta[c];
    Q[(size_t)r * D_ + c] = (bf16_t)(y * mk);
  }
}

// ---------------- K6: score GEMM, LDS-staged cb_key ------------------------
__global__ __launch_bounds__(256) void k_gemm_score(const bf16_t* __restrict__ Q,
                                                    const bf16_t* __restrict__ CK,
                                                    const float* __restrict__ lcov,
                                                    const float* __restrict__ maskc,
                                                    float* __restrict__ S) {
  __shared__ __align__(16) bf16_t sB[2][256 * 32];
  const int tid  = threadIdx.x;
  const int lane = tid & 31;
  const int wave = tid >> 5;
  const int rt   = wave >> 2;
  const int cg   = wave & 3;
  const int row0 = blockIdx.x * 32;
  const int cb   = blockIdx.y * 256;       // M=512 in two col-blocks
  const int b    = row0 / U_;
  const int m    = lane & 15;

  f32x8 acc[4] = {};
  const bf16_t* arow = Q + (size_t)(row0 + rt * 16 + m) * D_;

  auto prefetch = [&](int kt, int buf) {
    const bf16_t* src = CK + (size_t)(cb + tid) * D_ + kt * 32;
    uint32_t dst = (uint32_t)(uintptr_t)&sB[buf][tid * 32];
#pragma unroll
    for (int j = 0; j < 4; ++j) async_ld_b128(dst + j * 16, src + j * 8);
  };

  prefetch(0, 0);
  for (int kt = 0; kt < 8; ++kt) {
    wait_asynccnt0();
    __syncthreads();
    if (kt + 1 < 8) prefetch(kt + 1, (kt + 1) & 1);
    bf16x16 afrag = frag_load(arow, kt * 32, lane);
    const bf16_t* sbase = sB[kt & 1];
#pragma unroll
    for (int t = 0; t < 4; ++t) {
      bf16x16 bfrag = frag_load(sbase + (size_t)((cg * 4 + t) * 16 + m) * 32, 0, lane);
      acc[t] = __builtin_amdgcn_wmma_f32_16x16x32_bf16(
          false, afrag, false, bfrag, (short)0, acc[t], false, false);
    }
    __syncthreads();
  }
  const int rhi = (lane >> 4) * 8;
#pragma unroll
  for (int t = 0; t < 4; ++t) {
    int col = cb + (cg * 4 + t) * 16 + m;
    float lc = lcov[b * M_ + col];
#pragma unroll
    for (int v = 0; v < 8; ++v) {
      int r = row0 + rt * 16 + rhi + v;
      float sc = acc[t][v] * 0.0625f + lc;           // 1/sqrt(256)
      if (maskc[r] <= 0.0f) sc = -10000.0f;
      S[(size_t)r * M_ + col] = sc;
    }
  }
}

// ---------------- K7: fused softmax + attn.(value,var,cov) + extras --------
__global__ void k_softmax(const float* __restrict__ S, const float* __restrict__ rv,
                          const float* __restrict__ rva, const float* __restrict__ rc,
                          const float* __restrict__ grate, const float* __restrict__ lrs,
                          const float* __restrict__ speech, const float* __restrict__ maskc,
                          bf16_t* __restrict__ extras, float* __restrict__ gshift) {
  int lane = threadIdx.x & 31;
  int r = blockIdx.x * 8 + (threadIdx.x >> 5);
  int b = r / U_;
  const float* srow = S + (size_t)r * M_;
  float v[16];
#pragma unroll
  for (int j = 0; j < 16; ++j) v[j] = srow[lane + 32 * j];
  float mx = v[0];
#pragma unroll
  for (int j = 1; j < 16; ++j) mx = fmaxf(mx, v[j]);
  mx = wave_max(mx);
  float sum = 0.f, wv = 0.f, wva = 0.f, wc = 0.f;
#pragma unroll
  for (int j = 0; j < 16; ++j) {
    int mc = lane + 32 * j;
    float e = expf(v[j] - mx);
    sum += e;
    wv  += e * rv [b * M_ + mc];
    wva += e * rva[b * M_ + mc];
    wc  += e * rc [b * M_ + mc];
  }
  sum = wave_sum(sum); wv = wave_sum(wv); wva = wave_sum(wva); wc = wave_sum(wc);
  float mk    = maskc[r];
  float inv   = mk / sum;                        // attn = softmax * mask
  float rvu   = wv * inv;
  float rvaru = fmaxf(wva * inv, 1e-4f);
  float rcovu = fmaxf(wc * inv, 1e-4f);
  float conf  = rcovu / (rcovu + rvaru);
  float gs    = (grate[b] - lrs[r]) * speech[r];
  bf16_t ev = (bf16_t)0.0f;                      // 32-wide k-tile, zero padded
  if      (lane == 0) ev = (bf16_t)gs;
  else if (lane == 1) ev = (bf16_t)rvu;
  else if (lane == 2) ev = (bf16_t)conf;
  else if (lane == 3) ev = (bf16_t)logf(rvaru);
  extras[(size_t)r * 32 + lane] = ev;
  if (lane == 0) gshift[r] = gs;
}

// ---------------- K8: residual MLP GEMM + fused w2 reduce + final pred -----
__global__ void k_gemm_resid(const bf16_t* __restrict__ Q, const bf16_t* __restrict__ E,
                             const bf16_t* __restrict__ W1p, const float* __restrict__ b1,
                             const float* __restrict__ w2, const float* __restrict__ b2,
                             const float* __restrict__ gshift, const float* __restrict__ speech,
                             float* __restrict__ out) {
  __shared__ float rowsum[16];
  const int lane = threadIdx.x & 31;
  const int wave = threadIdx.x >> 5;
  const int row0 = blockIdx.x * 16;
  const int m    = lane & 15;

  f32x8 acc[4] = {};
  const bf16_t* arow = Q + (size_t)(row0 + m) * D_;
#pragma unroll
  for (int kt = 0; kt < 8; ++kt) {                // K 0..255: query part
    bf16x16 afrag = frag_load(arow, kt * 32, lane);
#pragma unroll
    for (int t = 0; t < 4; ++t) {
      const bf16_t* brow = W1p + (size_t)((wave * 4 + t) * 16 + m) * 288;
      bf16x16 bfrag = frag_load(brow, kt * 32, lane);
      acc[t] = __builtin_amdgcn_wmma_f32_16x16x32_bf16(
          false, afrag, false, bfrag, (short)0, acc[t], false, false);
    }
  }
  {                                               // K 256..287: extras tile
    bf16x16 afrag = frag_load(E + (size_t)(row0 + m) * 32, 0, lane);
#pragma unroll
    for (int t = 0; t < 4; ++t) {
      const bf16_t* brow = W1p + (size_t)((wave * 4 + t) * 16 + m) * 288;
      bf16x16 bfrag = frag_load(brow, 256, lane);
      acc[t] = __builtin_amdgcn_wmma_f32_16x16x32_bf16(
          false, afrag, false, bfrag, (short)0, acc[t], false, false);
    }
  }
  if (threadIdx.x < 16) rowsum[threadIdx.x] = 0.0f;
  __syncthreads();

  float part[8] = {};
#pragma unroll
  for (int t = 0; t < 4; ++t) {
    int col = (wave * 4 + t) * 16 + m;
    float w2c = w2[col], bb = b1[col];
#pragma unroll
    for (int v = 0; v < 8; ++v) part[v] += gelu_exact(acc[t][v] + bb) * w2c;
  }
  const int rhi = (lane >> 4) * 8;
#pragma unroll
  for (int v = 0; v < 8; ++v) atomicAdd(&rowsum[rhi + v], part[v]);
  __syncthreads();

  if (threadIdx.x < 16) {
    int r = row0 + threadIdx.x;
    float res = 0.35f * tanhf(rowsum[threadIdx.x] + b2[0]);
    float smv = speech[r];
    res *= smv;
    float p = gshift[r] + res;
    p = fminf(fmaxf(p, -MAXLS_), MAXLS_) * smv;
    out[r] = p;
  }
}

// ---------------------------------------------------------------------------
extern "C" void kernel_launch(void* const* d_in, const int* in_sizes, int n_in,
                              void* d_out, int out_size, void* d_ws, size_t ws_size,
                              hipStream_t stream) {
  const int*   content     = (const int*)  d_in[0];
  const float* log_anchor  = (const float*)d_in[1];
  const float* unit_mask   = (const float*)d_in[2];
  const float* sealed_mask = (const float*)d_in[3];
  const float* sep_hint    = (const float*)d_in[4];
  const float* global_rate = (const float*)d_in[6];
  const float* role_value  = (const float*)d_in[7];
  const float* role_var    = (const float*)d_in[8];
  const float* role_cov    = (const float*)d_in[9];
  const float* lre         = (const float*)d_in[10];
  const float* emb         = (const float*)d_in[11];
  const float* aux_w       = (const float*)d_in[12];
  const float* aux_b       = (const float*)d_in[13];
  const float* conv1_w     = (const float*)d_in[14];
  const float* conv1_b     = (const float*)d_in[15];
  const float* conv2_w     = (const float*)d_in[16];
  const float* conv2_b     = (const float*)d_in[17];
  const float* ln_g        = (const float*)d_in[18];
  const float* ln_b        = (const float*)d_in[19];
  const float* cb_key      = (const float*)d_in[20];
  const float* rh_w1       = (const float*)d_in[21];
  const float* rh_b1       = (const float*)d_in[22];
  const float* rh_w2       = (const float*)d_in[23];
  const float* rh_b2       = (const float*)d_in[24];
  float* out = (float*)d_out;

  char* ws = (char*)d_ws;
  size_t off = 0;
  auto alloc = [&](size_t bytes) { char* p = ws + off; off += (bytes + 255) & ~(size_t)255; return p; };

  float*  maskc  = (float*) alloc((size_t)R_ * 4);
  float*  speech = (float*) alloc((size_t)R_ * 4);
  float*  lrs    = (float*) alloc((size_t)R_ * 4);
  float*  gshift = (float*) alloc((size_t)R_ * 4);
  bf16_t* h0     = (bf16_t*)alloc((size_t)R_ * D_ * 2);   // reused for conv2 out
  bf16_t* c1o    = (bf16_t*)alloc((size_t)R_ * D_ * 2);
  bf16_t* query  = (bf16_t*)alloc((size_t)R_ * D_ * 2);
  float*  score  = (float*) alloc((size_t)R_ * M_ * 4);
  bf16_t* extras = (bf16_t*)alloc((size_t)R_ * 32 * 2);
  bf16_t* p1     = (bf16_t*)alloc((size_t)D_ * D_ * 3 * 2);
  bf16_t* p2     = (bf16_t*)alloc((size_t)D_ * D_ * 3 * 2);
  bf16_t* pck    = (bf16_t*)alloc((size_t)M_ * D_ * 2);
  bf16_t* pw1    = (bf16_t*)alloc((size_t)D_ * 288 * 2);
  float*  lcov   = (float*) alloc((size_t)B_ * M_ * 4);

  k_prepack<<<768, 256, 0, stream>>>(conv1_w, conv2_w, cb_key, rh_w1, role_cov,
                                     p1, p2, pck, pw1, lcov);
  k_scan<<<4, 256, 0, stream>>>(log_anchor, unit_mask, sealed_mask, sep_hint,
                                lre, maskc, speech, lrs);
  k_hidden0<<<R_ / 4, 256, 0, stream>>>(content, log_anchor, lrs, emb, aux_w, aux_b, h0);
  k_gemm_conv<3><<<R_ / 32, 256, 0, stream>>>(h0, p1, conv1_b, c1o);
  k_gemm_conv<3><<<R_ / 32, 256, 0, stream>>>(c1o, p2, conv2_b, h0);
  k_layernorm<<<R_ / 8, 256, 0, stream>>>(h0, ln_g, ln_b, maskc, query);
  k_gemm_score<<<dim3(R_ / 32, 2), 256, 0, stream>>>(query, pck, lcov, maskc, score);
  k_softmax<<<R_ / 8, 256, 0, stream>>>(score, role_value, role_var, role_cov,
                                        global_rate, lrs, speech, maskc, extras, gshift);
  k_gemm_resid<<<R_ / 16, 128, 0, stream>>>(query, extras, pw1, rh_b1, rh_w2, rh_b2,
                                            gshift, speech, out);
}
